// TransformerBlock_45689862095520
// MI455X (gfx1250) — matrix operand order repaired and checked by reading, not compiled, not run on hardware
//
#include <hip/hip_runtime.h>
#include <hip/hip_bf16.h>

// ---------------- fixed problem sizes ----------------
#define DM     768
#define NH     12
#define DH     64
#define T_SEQ  4096
#define NB     2
#define M_TOK  (NB * T_SEQ)   // 8192
#define FF     3072
#define EPSV   1e-6f

// ---------------- WMMA vector types ----------------
typedef __attribute__((ext_vector_type(16))) __bf16 v16bf;
typedef __attribute__((ext_vector_type(8)))  __bf16 v8bf;
typedef __attribute__((ext_vector_type(8)))  float  v8f;

// bf16 <-> f32 via bit ops (storage-only use of __bf16, RNE rounding)
static __device__ __forceinline__ __bf16 f2b(float f) {
  unsigned u = __float_as_uint(f);
  u += 0x7FFFu + ((u >> 16) & 1u);
  unsigned short s = (unsigned short)(u >> 16);
  __bf16 r;
  __builtin_memcpy(&r, &s, 2);
  return r;
}

static __device__ __forceinline__ v8f wmma_bf16(v16bf a, v16bf b, v8f c) {
  return __builtin_amdgcn_wmma_f32_16x16x32_bf16(false, a, false, b, (short)0, c, false, false);
}

// A-layout within a 32-wide K chunk: identity (contiguous K order, matching
// B-fragments: lane half h holds K 16h..16h+15 at offset half*16).
static __device__ __forceinline__ int aperm(int kk) {
  return kk;
}

// CDNA5 async copy: global memory -> LDS, 16 bytes per lane, ASYNCcnt-tracked.
// ldsa = workgroup-relative LDS byte address (generic-pointer low 32 bits),
// ga   = 64-bit global address.
static __device__ __forceinline__ void async_g2l_b128(unsigned ldsa, unsigned long long ga) {
  asm volatile("global_load_async_to_lds_b128 %0, %1, off"
               :: "v"(ldsa), "v"(ga) : "memory");
}
static __device__ __forceinline__ void wait_async0() {
  asm volatile("s_wait_asynccnt 0" ::: "memory");
}
static __device__ __forceinline__ unsigned lds_addr32(const void* p) {
  return (unsigned)(unsigned long long)p;   // LDS aperture keeps offset in addr[31:0]
}

// ---------------- fp32 -> bf16 conversion ----------------
__global__ void cvt_bf16_kernel(const float* __restrict__ in, __bf16* __restrict__ out, int n) {
  int i = blockIdx.x * blockDim.x + threadIdx.x;
  if (i < n) out[i] = f2b(in[i]);
}

// ---------------- RMSNorm: one block per row of 768 ----------------
__global__ __launch_bounds__(256) void rmsnorm_kernel(const float* __restrict__ x,
                                                      const float* __restrict__ g,
                                                      __bf16* __restrict__ out) {
  __shared__ float red[8];
  const int row = blockIdx.x;
  const float* xr = x + (size_t)row * DM;
  const int t = threadIdx.x;
  float v0 = xr[t], v1 = xr[t + 256], v2 = xr[t + 512];
  float s = v0 * v0 + v1 * v1 + v2 * v2;
  #pragma unroll
  for (int m = 1; m < 32; m <<= 1) s += __shfl_xor(s, m, 32);
  if ((t & 31) == 0) red[t >> 5] = s;
  __syncthreads();
  if (t == 0) {
    float tot = 0.f;
    #pragma unroll
    for (int i = 0; i < 8; i++) tot += red[i];
    red[0] = rsqrtf(tot / (float)DM + EPSV);
  }
  __syncthreads();
  const float rms = red[0];
  __bf16* orow = out + (size_t)row * DM;
  orow[t]       = f2b(v0 * rms * g[t]);
  orow[t + 256] = f2b(v1 * rms * g[t + 256]);
  orow[t + 512] = f2b(v2 * rms * g[t + 512]);
}

// ---------------- tiled bf16 WMMA GEMM ----------------
// C(MxN) = A(MxK,bf16 row-major) * B(KxN,bf16 row-major), fused epilogue.
// 128 threads (4 waves). Block tile 64x128, wave tile 32x64, BK=32.
// Double-buffered LDS; A staged by async global->LDS DMA in fragment-ready
// natural layout; B staged through registers (transpose).
#define BM 64
#define BN 128
#define BK 32

enum { EPI_QA = 0, EPI_KB = 1, EPI_VT = 2, EPI_GELU = 3, EPI_RES = 4 };

template <int EPI>
__global__ __launch_bounds__(128) void gemm_bf16_kernel(
    const __bf16* __restrict__ A, const __bf16* __restrict__ B,
    const float* __restrict__ bias, const float* __restrict__ res,
    __bf16* __restrict__ outB, float* __restrict__ outF,
    int M, int N, int K) {
  __shared__ __align__(32) __bf16 As[2][BM][BK];   // A tiles, natural K order
  __shared__ __align__(32) __bf16 Bt[2][BN][BK];   // transposed B: Bt[n][k]

  const int tid  = threadIdx.x;
  const int lane = tid & 31;
  const int wid  = tid >> 5;
  const int half = lane >> 4;
  const int l16  = lane & 15;
  const int wr   = (wid >> 1) * 32;   // wave rows in block tile
  const int wc   = (wid & 1) * 64;    // wave cols in block tile
  const int bm   = blockIdx.x * BM;
  const int bn   = blockIdx.y * BN;

  // loader assignments
  const int arow = tid >> 1;           // 0..63
  const int acol = (tid & 1) * 16;     // 0 or 16
  const int ad0  = acol;               // natural dest: 0 or 16
  const int bkk  = tid & 31;           // 0..31
  const int bn0  = (tid >> 5) * 16;    // 0,16,32,48

  v8f acc[2][4] = {};
  const int nkt = K / BK;

  auto stage = [&](int kt, int buf) {
    // A tile: async DMA global -> LDS (two 16B chunks, natural destinations)
    const __bf16* asrc = A + (size_t)(bm + arow) * K + kt * BK + acol;
    async_g2l_b128(lds_addr32(&As[buf][arow][ad0]),      (unsigned long long)asrc);
    async_g2l_b128(lds_addr32(&As[buf][arow][ad0 + 8]),  (unsigned long long)(asrc + 8));
    // B tile: vector global load, scattered transpose store
    #pragma unroll
    for (int g = 0; g < 2; ++g) {
      const __bf16* bsrc = B + (size_t)(kt * BK + bkk) * N + bn + g * 64 + bn0;
      v8bf t0 = *(const v8bf*)bsrc;
      v8bf t1 = *(const v8bf*)(bsrc + 8);
      #pragma unroll
      for (int i = 0; i < 8; i++) {
        Bt[buf][g * 64 + bn0 + i][bkk]     = t0[i];
        Bt[buf][g * 64 + bn0 + 8 + i][bkk] = t1[i];
      }
    }
    // prefetch tile after next into L2
    if (kt + 1 < nkt) {
      __builtin_prefetch(A + (size_t)(bm + arow) * K + (kt + 1) * BK + acol, 0, 1);
      __builtin_prefetch(B + (size_t)((kt + 1) * BK + bkk) * N + bn + bn0, 0, 1);
    }
  };

  stage(0, 0);
  wait_async0();          // A-tile DMA complete before barrier
  __syncthreads();

  for (int kt = 0; kt < nkt; ++kt) {
    const int buf = kt & 1;
    if (kt + 1 < nkt) stage(kt + 1, buf ^ 1);   // overlap next-tile loads with math

    v16bf aF[2], bF[4];
    #pragma unroll
    for (int i = 0; i < 2; i++)
      aF[i] = *(const v16bf*)&As[buf][wr + i * 16 + l16][half * 16];
    #pragma unroll
    for (int j = 0; j < 4; j++)
      bF[j] = *(const v16bf*)&Bt[buf][wc + j * 16 + l16][half * 16];

    #pragma unroll
    for (int i = 0; i < 2; i++)
      #pragma unroll
      for (int j = 0; j < 4; j++)
        acc[i][j] = wmma_bf16(aF[i], bF[j], acc[i][j]);

    wait_async0();        // next A tile landed in LDS
    __syncthreads();
  }

  // epilogue: C layout -> lane col n=l16, rows m = half*8 + r
  #pragma unroll
  for (int i = 0; i < 2; i++)
    #pragma unroll
    for (int j = 0; j < 4; j++)
      #pragma unroll
      for (int r = 0; r < 8; r++) {
        const int gm = bm + wr + i * 16 + half * 8 + r;
        const int gn = bn + wc + j * 16 + l16;
        const float v = acc[i][j][r];
        if constexpr (EPI == EPI_QA) {
          // Q: [B][H][T][64], natural dh order (read as contiguous A-fragments)
          const int b = gm >> 12, t = gm & (T_SEQ - 1);
          const int h = gn >> 6, dh = gn & 63;
          const int c = dh >> 5, jj = aperm(dh & 31);
          outB[(((size_t)(b * NH + h)) * T_SEQ + t) * DH + c * 32 + jj] = f2b(v);
        } else if constexpr (EPI == EPI_KB) {
          // K: [B][H][T][64] natural (read as contiguous B-fragments)
          const int b = gm >> 12, t = gm & (T_SEQ - 1);
          const int h = gn >> 6, dh = gn & 63;
          outB[(((size_t)(b * NH + h)) * T_SEQ + t) * DH + dh] = f2b(v);
        } else if constexpr (EPI == EPI_VT) {
          // V^T: [B][H][64][T] (contiguous B-fragments over keys)
          const int b = gm >> 12, t = gm & (T_SEQ - 1);
          const int h = gn >> 6, dh = gn & 63;
          outB[((size_t)(b * NH + h) * DH + dh) * T_SEQ + t] = f2b(v);
        } else if constexpr (EPI == EPI_GELU) {
          const float xg = v + bias[gn];
          const float gl = 0.5f * xg * (1.f + erff(xg * 0.70710678118654752f));
          outB[(size_t)gm * N + gn] = f2b(gl);
        } else {  // EPI_RES
          float xo = v + res[(size_t)gm * N + gn];
          if (bias) xo += bias[gn];
          outF[(size_t)gm * N + gn] = xo;
        }
      }
}

// ---------------- flash attention ----------------
// q, k : [B][H][T][64] bf16 ; vt : [B][H][64][T] bf16 ;
// out : [B][T][768] bf16. One wave per 16-query tile; 4 waves per block.
__global__ __launch_bounds__(128) void attn_kernel(const __bf16* __restrict__ q,
                                                   const __bf16* __restrict__ k,
                                                   const __bf16* __restrict__ vt,
                                                   __bf16* __restrict__ out) {
  __shared__ __align__(32) __bf16 Plds[4][16 * 32];  // per-wave P staging (natural key order)

  const int lane = threadIdx.x & 31;
  const int wid  = threadIdx.x >> 5;
  const int gw   = blockIdx.x * 4 + wid;
  const int nqt  = T_SEQ / 16;            // 256
  const int qt = gw % nqt;
  const int h  = (gw / nqt) % NH;
  const int b  = gw / (nqt * NH);
  const int qBase = qt * 16;
  const int half = lane >> 4;
  const int l16  = lane & 15;

  const __bf16* qp = q  + (size_t)(b * NH + h) * T_SEQ * DH;
  const __bf16* kp = k  + (size_t)(b * NH + h) * T_SEQ * DH;
  const __bf16* vp = vt + (size_t)(b * NH + h) * DH * T_SEQ;

  // Q fragments (A layout, natural order in memory -> single contiguous v16bf)
  v16bf qF[2];
  #pragma unroll
  for (int c = 0; c < 2; c++)
    qF[c] = *(const v16bf*)(qp + (size_t)(qBase + l16) * DH + c * 32 + half * 16);

  float mrow[8], lrow[8];
  v8f oacc[4] = {};
  #pragma unroll
  for (int r = 0; r < 8; r++) { mrow[r] = -1e30f; lrow[r] = 0.f; }

  const int nkt = (qBase + 16 + 31) / 32;  // causal: keys <= qBase+15
  for (int kt = 0; kt < nkt; ++kt) {
    const int kb = kt * 32;
    float sv[2][8];
    #pragma unroll
    for (int sub = 0; sub < 2; ++sub) {
      const int keyB = kb + sub * 16;
      // K^T B-fragments: lane col = key, elems = contiguous Dh (32B v16bf)
      v16bf kF0 = *(const v16bf*)(kp + (size_t)(keyB + l16) * DH + 0 * 32 + half * 16);
      v16bf kF1 = *(const v16bf*)(kp + (size_t)(keyB + l16) * DH + 1 * 32 + half * 16);
      v8f s = {};
      s = wmma_bf16(qF[0], kF0, s);
      s = wmma_bf16(qF[1], kF1, s);
      const int key = keyB + l16;
      #pragma unroll
      for (int r = 0; r < 8; r++) {
        const int qi = qBase + half * 8 + r;
        const float xs = s[r] * 0.125f;               // Dh^-0.5 = 1/8
        sv[sub][r] = (key > qi) ? -1e30f : xs;
      }
    }
    // online softmax: per-row reductions across the 16-lane half-group
    #pragma unroll
    for (int r = 0; r < 8; r++) {
      float mx = fmaxf(sv[0][r], sv[1][r]);
      #pragma unroll
      for (int m = 1; m < 16; m <<= 1) mx = fmaxf(mx, __shfl_xor(mx, m, 32));
      const float mnew = fmaxf(mrow[r], mx);
      const float sf = __expf(mrow[r] - mnew);
      mrow[r] = mnew;
      lrow[r] *= sf;
      #pragma unroll
      for (int c = 0; c < 4; c++) oacc[c][r] *= sf;
      const float p0 = __expf(sv[0][r] - mnew);
      const float p1 = __expf(sv[1][r] - mnew);
      float rs = p0 + p1;
      #pragma unroll
      for (int m = 1; m < 16; m <<= 1) rs += __shfl_xor(rs, m, 32);
      lrow[r] += rs;
      const int mi = half * 8 + r;
      Plds[wid][mi * 32 + aperm(l16)]      = f2b(p0);   // keys 0..15 natural order
      Plds[wid][mi * 32 + aperm(16 + l16)] = f2b(p1);   // keys 16..31 natural order
    }
    asm volatile("s_wait_dscnt 0" ::: "memory");  // same-wave LDS RAW fence

    // P A-fragment: natural layout -> single contiguous v16bf per lane
    v16bf pF = *(const v16bf*)&Plds[wid][l16 * 32 + half * 16];

    // P * V : V^T B-fragments contiguous (vt is [Dh][T])
    #pragma unroll
    for (int c = 0; c < 4; c++) {
      const int dh = c * 16 + l16;
      v16bf vF = *(const v16bf*)(vp + (size_t)dh * T_SEQ + kb + half * 16);
      oacc[c] = wmma_bf16(pF, vF, oacc[c]);
    }
  }

  // write out: [B][T][H*64+dh] bf16
  __bf16* op = out + (size_t)b * T_SEQ * DM + h * DH;
  #pragma unroll
  for (int c = 0; c < 4; c++)
    #pragma unroll
    for (int r = 0; r < 8; r++) {
      const int t  = qBase + half * 8 + r;
      const int dh = c * 16 + l16;
      op[(size_t)t * DM + dh] = f2b(oacc[c][r] / lrow[r]);
    }
}

// ---------------- launch ----------------
extern "C" void kernel_launch(void* const* d_in, const int* in_sizes, int n_in,
                              void* d_out, int out_size, void* d_ws, size_t ws_size,
                              hipStream_t stream) {
  const float* x  = (const float*)d_in[0];
  const float* wq = (const float*)d_in[1];
  const float* wk = (const float*)d_in[2];
  const float* wv = (const float*)d_in[3];
  const float* wo = (const float*)d_in[4];
  const float* w1 = (const float*)d_in[5];
  const float* b1 = (const float*)d_in[6];
  const float* w2 = (const float*)d_in[7];
  const float* b2 = (const float*)d_in[8];
  const float* g1 = (const float*)d_in[9];
  const float* g2 = (const float*)d_in[10];

  char* ws = (char*)d_ws;
  size_t off = 0;
  auto give = [&](size_t bytes) -> char* {
    char* p = ws + off;
    off += (bytes + 255) & ~(size_t)255;
    return p;
  };

  __bf16* wq_b = (__bf16*)give((size_t)DM * DM * 2);
  __bf16* wk_b = (__bf16*)give((size_t)DM * DM * 2);
  __bf16* wv_b = (__bf16*)give((size_t)DM * DM * 2);
  __bf16* wo_b = (__bf16*)give((size_t)DM * DM * 2);
  __bf16* w1_b = (__bf16*)give((size_t)DM * FF * 2);
  __bf16* w2_b = (__bf16*)give((size_t)FF * DM * 2);
  __bf16* xn1  = (__bf16*)give((size_t)M_TOK * DM * 2);
  __bf16* qb   = (__bf16*)give((size_t)M_TOK * DM * 2);   // [B][H][T][Dh]
  __bf16* kb   = (__bf16*)give((size_t)M_TOK * DM * 2);   // [B][H][T][Dh]
  __bf16* vtb  = (__bf16*)give((size_t)M_TOK * DM * 2);   // [B][H][Dh][T]
  __bf16* ao   = (__bf16*)give((size_t)M_TOK * DM * 2);   // attention out bf16
  float*  x1   = (float*) give((size_t)M_TOK * DM * 4);   // residual 1 fp32
  __bf16* xn2  = (__bf16*)give((size_t)M_TOK * DM * 2);
  __bf16* hb   = (__bf16*)give((size_t)M_TOK * FF * 2);   // FFN hidden bf16
  (void)ws_size; (void)in_sizes; (void)n_in; (void)out_size;

  // weight conversions
  {
    const int n1 = DM * DM, n2 = DM * FF;
    cvt_bf16_kernel<<<(n1 + 255) / 256, 256, 0, stream>>>(wq, wq_b, n1);
    cvt_bf16_kernel<<<(n1 + 255) / 256, 256, 0, stream>>>(wk, wk_b, n1);
    cvt_bf16_kernel<<<(n1 + 255) / 256, 256, 0, stream>>>(wv, wv_b, n1);
    cvt_bf16_kernel<<<(n1 + 255) / 256, 256, 0, stream>>>(wo, wo_b, n1);
    cvt_bf16_kernel<<<(n2 + 255) / 256, 256, 0, stream>>>(w1, w1_b, n2);
    cvt_bf16_kernel<<<(n2 + 255) / 256, 256, 0, stream>>>(w2, w2_b, n2);
  }

  // 1) xn1 = rmsnorm(x, g1)
  rmsnorm_kernel<<<M_TOK, 256, 0, stream>>>(x, g1, xn1);

  // 2) Q,K,V projections (WMMA GEMMs with layout epilogues)
  dim3 gQKV(M_TOK / BM, DM / BN);
  gemm_bf16_kernel<EPI_QA><<<gQKV, 128, 0, stream>>>(xn1, wq_b, nullptr, nullptr, qb,  nullptr, M_TOK, DM, DM);
  gemm_bf16_kernel<EPI_KB><<<gQKV, 128, 0, stream>>>(xn1, wk_b, nullptr, nullptr, kb,  nullptr, M_TOK, DM, DM);
  gemm_bf16_kernel<EPI_VT><<<gQKV, 128, 0, stream>>>(xn1, wv_b, nullptr, nullptr, vtb, nullptr, M_TOK, DM, DM);

  // 3) flash attention: one wave per 16-query tile
  {
    const int total_waves = NB * NH * (T_SEQ / 16);   // 6144
    attn_kernel<<<total_waves / 4, 128, 0, stream>>>(qb, kb, vtb, ao);
  }

  // 4) output projection + residual: x1 = x + ao @ wo
  gemm_bf16_kernel<EPI_RES><<<gQKV, 128, 0, stream>>>(ao, wo_b, nullptr, x, nullptr, x1, M_TOK, DM, DM);

  // 5) xn2 = rmsnorm(x1, g2)
  rmsnorm_kernel<<<M_TOK, 256, 0, stream>>>(x1, g2, xn2);

  // 6) FFN up + GELU: h = gelu(xn2 @ w1 + b1)
  dim3 gUP(M_TOK / BM, FF / BN);
  gemm_bf16_kernel<EPI_GELU><<<gUP, 128, 0, stream>>>(xn2, w1_b, b1, nullptr, hb, nullptr, M_TOK, FF, DM);

  // 7) FFN down + bias + residual: out = x1 + h @ w2 + b2
  gemm_bf16_kernel<EPI_RES><<<gQKV, 128, 0, stream>>>(hb, w2_b, b2, x1, nullptr, (float*)d_out, M_TOK, DM, FF);
}
